// AdaptiveFusionLayer_901943132283
// MI455X (gfx1250) — matrix operand order repaired and checked by reading, not compiled, not run on hardware
//
#include <hip/hip_runtime.h>
#include <hip/hip_bf16.h>
#include <stdint.h>

// ---------------------------------------------------------------------------
// AdaptiveFusionLayer for MI455X (gfx1250, wave32, WMMA bf16 w/ f32 accum)
// ---------------------------------------------------------------------------

typedef __attribute__((ext_vector_type(16))) __bf16 v16bf;
typedef __attribute__((ext_vector_type(8)))  __bf16 v8bf;
typedef __attribute__((ext_vector_type(8)))  float  v8f;

static constexpr int N_ROWS    = 50000;
static constexpr int D         = 768;
static constexpr int BM        = 64;        // rows per workgroup in main kernel
static constexpr int LDS_STR   = D + 8;     // bf16 elems per LDS row (pad: 4-bank skew)
static constexpr int KTILES    = D / 32;    // 24
static constexpr int NTILES    = D / 16;    // 48

__device__ __forceinline__ uint32_t f32_to_bf16_rne(float f) {
    uint32_t u = __float_as_uint(f);
    return (u + 0x7FFFu + ((u >> 16) & 1u)) >> 16;   // round-to-nearest-even
}

// ---------------------------------------------------------------------------
// Kernel 1: v2[i] = sum_j W[i][j] * a[D + j]     (W @ a2), D floats out
// ---------------------------------------------------------------------------
__global__ void v2_kernel(const float* __restrict__ W,
                          const float* __restrict__ a,
                          float* __restrict__ v2) {
    int i = blockIdx.x * blockDim.x + threadIdx.x;
    if (i >= D) return;
    const float* row = W + (size_t)i * D;
    const float* a2  = a + D;
    float acc = 0.f;
    #pragma unroll 4
    for (int j = 0; j < D; j += 4) {
        float4 w4 = *(const float4*)(row + j);
        float4 x4 = *(const float4*)(a2 + j);
        acc += w4.x * x4.x + w4.y * x4.y + w4.z * x4.z + w4.w * x4.w;
    }
    v2[i] = acc;
}

// ---------------------------------------------------------------------------
// Kernel 2: pack W (f32 row-major KxN) into WMMA B-operand tiles (bf16).
// Tile (nt,kt) = 32x16 block of W; 1024 bytes: dword index = lane*8 + v.
// Lane n in 0..15 holds column nt*16+n; VGPR v packs K = kt*32 + 2v,2v+1;
// lanes 16..31 hold the same columns at K offset +16.  (contiguous k order, same as A operand)
// ---------------------------------------------------------------------------
__global__ void packW_kernel(const float* __restrict__ W,
                             uint32_t* __restrict__ Wp) {
    int t      = blockIdx.x * blockDim.x + threadIdx.x;   // one packed dword
    int tile   = t >> 8;                                  // 256 dwords / tile
    int within = t & 255;
    int lane   = within >> 3;
    int v      = within & 7;
    int nt     = tile / KTILES;
    int kt     = tile % KTILES;
    int col    = nt * 16 + (lane & 15);
    int k0     = kt * 32 + ((lane >> 4) << 4) + 2 * v;
    uint32_t lo = f32_to_bf16_rne(W[(size_t)k0 * D + col]);
    uint32_t hi = f32_to_bf16_rne(W[(size_t)(k0 + 1) * D + col]);
    Wp[t] = lo | (hi << 16);
}

// ---------------------------------------------------------------------------
// Kernel 3: attention logits + softmax.  One wave32 per node n.
// logit k = F_{1 + r/N}[r%N] . v2  with r = 3n+k  (faithful block reshape).
// Wh1 is constant across k -> cancels in softmax; omitted.
// ---------------------------------------------------------------------------
__global__ void attn_e_kernel(const float* __restrict__ F1,
                              const float* __restrict__ F2,
                              const float* __restrict__ F3,
                              const float* __restrict__ v2,
                              float* __restrict__ e_out) {
    const int lane = threadIdx.x & 31;
    const int wave = threadIdx.x >> 5;
    const int n    = blockIdx.x * 8 + wave;
    if (n >= N_ROWS) return;

    float g[3];
    #pragma unroll
    for (int k = 0; k < 3; ++k) {
        int r   = 3 * n + k;
        int src = r / N_ROWS;
        int row = r - src * N_ROWS;
        const float* F = (src == 0) ? F1 : ((src == 1) ? F2 : F3);
        F += (size_t)row * D;
        float acc = 0.f;
        #pragma unroll
        for (int i = 0; i < D / 128; ++i) {          // 6 iters, 128 f32 / iter / wave
            int j = i * 128 + lane * 4;
            float4 f = *(const float4*)(F + j);
            float4 w = *(const float4*)(v2 + j);
            acc += f.x * w.x + f.y * w.y + f.z * w.z + f.w * w.w;
        }
        #pragma unroll
        for (int off = 16; off > 0; off >>= 1) acc += __shfl_xor(acc, off, 32);
        g[k] = acc;                                   // uniform across wave
    }
    float m  = fmaxf(g[0], fmaxf(g[1], g[2]));
    float e0 = __expf(g[0] - m), e1 = __expf(g[1] - m), e2 = __expf(g[2] - m);
    float inv = 1.f / (e0 + e1 + e2);
    if (lane == 0) {
        float* dst = e_out + (size_t)n * 3;
        dst[0] = e0 * inv; dst[1] = e1 * inv; dst[2] = e2 * inv;
    }
}

// ---------------------------------------------------------------------------
// Kernel 4: out = relu( (e0*F1 + e1*F2 + e2*F3) @ W ) + F0
// Phase A: build wh (bf16) in LDS, row-major, padded stride.
// Phase B: 8 waves; wave w -> rows (w&3)*16..+16, cols (w>>2)*384..+384,
//          6 chunks of 64 cols, K loop of 24 x v_wmma_f32_16x16x32_bf16 x4.
// ---------------------------------------------------------------------------
__global__ __launch_bounds__(256) void fused_wh_gemm_kernel(
        const float* __restrict__ F0, const float* __restrict__ F1,
        const float* __restrict__ F2, const float* __restrict__ F3,
        const float* __restrict__ e,  const uint32_t* __restrict__ Wp,
        float* __restrict__ out) {
    extern __shared__ char smem[];
    uint16_t* whs = (uint16_t*)smem;                      // [BM][LDS_STR] bf16
    float*    es  = (float*)(smem + (size_t)BM * LDS_STR * 2);  // [BM][3]

    const int tid     = threadIdx.x;
    const int rowBase = blockIdx.x * BM;

    if (tid < BM * 3) {
        int m = tid / 3, k = tid - 3 * m;
        int row = rowBase + m;
        es[tid] = (row < N_ROWS) ? e[(size_t)row * 3 + k] : 0.f;
    }
    __syncthreads();

    // ---- Phase A: wh -> LDS (bf16), zeros for out-of-range rows ----
    #pragma unroll
    for (int it = 0; it < (BM * (D / 4)) / 256; ++it) {   // 48 iters
        int idx = it * 256 + tid;
        int m   = idx / (D / 4);
        int d   = (idx - m * (D / 4)) * 4;
        int row = rowBase + m;
        float4 acc = {0.f, 0.f, 0.f, 0.f};
        if (row < N_ROWS) {
            size_t off = (size_t)row * D + d;
            float e0 = es[m * 3 + 0], e1 = es[m * 3 + 1], e2 = es[m * 3 + 2];
            float4 fa = *(const float4*)(F1 + off);
            float4 fb = *(const float4*)(F2 + off);
            float4 fc = *(const float4*)(F3 + off);
            acc.x = e0 * fa.x + e1 * fb.x + e2 * fc.x;
            acc.y = e0 * fa.y + e1 * fb.y + e2 * fc.y;
            acc.z = e0 * fa.z + e1 * fb.z + e2 * fc.z;
            acc.w = e0 * fa.w + e1 * fb.w + e2 * fc.w;
        }
        uint32_t p0 = f32_to_bf16_rne(acc.x) | (f32_to_bf16_rne(acc.y) << 16);
        uint32_t p1 = f32_to_bf16_rne(acc.z) | (f32_to_bf16_rne(acc.w) << 16);
        *(uint2*)(whs + (size_t)m * LDS_STR + d) = make_uint2(p0, p1);
    }
    __syncthreads();

    // ---- Phase B: WMMA GEMM ----
    const int lane    = tid & 31;
    const int wave    = tid >> 5;
    const int mTile   = wave & 3;
    const int colHalf = wave >> 2;
    // A-operand addressing (16-bit A 16x32, contiguous k order as in packed W):
// lane<16: row=lane,   K+0..15
    // lane>=16: row=lane-16, K+16..31
    const uint16_t* aPtr = whs + (size_t)(mTile * 16 + (lane & 15)) * LDS_STR
                               + ((lane >> 4) << 4);

    for (int chunk = 0; chunk < 6; ++chunk) {
        const int colBase = colHalf * 384 + chunk * 64;
        const int nt0     = colBase >> 4;
        v8f c0 = {}, c1 = {}, c2 = {}, c3 = {};

        for (int kt = 0; kt < KTILES; ++kt) {
            v8bf alo = *(const v8bf*)(aPtr + kt * 32);
            v8bf ahi = *(const v8bf*)(aPtr + kt * 32 + 8);
            v16bf A = __builtin_shufflevector(alo, ahi,
                        0,1,2,3,4,5,6,7,8,9,10,11,12,13,14,15);

            const uint32_t* bp0 = Wp + (((size_t)(nt0 + 0) * KTILES + kt) << 8) + lane * 8;
            const uint32_t* bp1 = Wp + (((size_t)(nt0 + 1) * KTILES + kt) << 8) + lane * 8;
            const uint32_t* bp2 = Wp + (((size_t)(nt0 + 2) * KTILES + kt) << 8) + lane * 8;
            const uint32_t* bp3 = Wp + (((size_t)(nt0 + 3) * KTILES + kt) << 8) + lane * 8;
            v16bf B0 = __builtin_shufflevector(*(const v8bf*)bp0, *(const v8bf*)(bp0 + 4),
                        0,1,2,3,4,5,6,7,8,9,10,11,12,13,14,15);
            v16bf B1 = __builtin_shufflevector(*(const v8bf*)bp1, *(const v8bf*)(bp1 + 4),
                        0,1,2,3,4,5,6,7,8,9,10,11,12,13,14,15);
            v16bf B2 = __builtin_shufflevector(*(const v8bf*)bp2, *(const v8bf*)(bp2 + 4),
                        0,1,2,3,4,5,6,7,8,9,10,11,12,13,14,15);
            v16bf B3 = __builtin_shufflevector(*(const v8bf*)bp3, *(const v8bf*)(bp3 + 4),
                        0,1,2,3,4,5,6,7,8,9,10,11,12,13,14,15);

            c0 = __builtin_amdgcn_wmma_f32_16x16x32_bf16(false, A, false, B0, (short)0, c0, false, false);
            c1 = __builtin_amdgcn_wmma_f32_16x16x32_bf16(false, A, false, B1, (short)0, c1, false, false);
            c2 = __builtin_amdgcn_wmma_f32_16x16x32_bf16(false, A, false, B2, (short)0, c2, false, false);
            c3 = __builtin_amdgcn_wmma_f32_16x16x32_bf16(false, A, false, B3, (short)0, c3, false, false);
        }

        // Epilogue: C layout = VGPR r, lanes 0-15 -> (M=r, N=lane),
        // lanes 16-31 -> (M=r+8, N=lane-16). relu + residual, coalesced.
        const int col    = colBase + (lane & 15);
        const int rowTop = rowBase + mTile * 16 + ((lane >> 4) << 3);
        #pragma unroll
        for (int r = 0; r < 8; ++r) {
            int row = rowTop + r;
            if (row < N_ROWS) {
                size_t o = (size_t)row * D + col;
                out[o]      = fmaxf(c0[r], 0.f) + F0[o];
                out[o + 16] = fmaxf(c1[r], 0.f) + F0[o + 16];
                out[o + 32] = fmaxf(c2[r], 0.f) + F0[o + 32];
                out[o + 48] = fmaxf(c3[r], 0.f) + F0[o + 48];
            }
        }
    }
}

// ---------------------------------------------------------------------------
extern "C" void kernel_launch(void* const* d_in, const int* in_sizes, int n_in,
                              void* d_out, int out_size, void* d_ws, size_t ws_size,
                              hipStream_t stream) {
    const float* F0 = (const float*)d_in[0];
    const float* F1 = (const float*)d_in[1];
    const float* F2 = (const float*)d_in[2];
    const float* F3 = (const float*)d_in[3];
    const float* W  = (const float*)d_in[4];
    const float* a  = (const float*)d_in[5];
    float* out = (float*)d_out;

    // Workspace layout (~1.79 MB total)
    char* ws = (char*)d_ws;
    float*    v2 = (float*)ws;                                  // 3 KB
    float*    e  = (float*)(ws + 4096);                         // N*3*4 = 600 KB
    uint32_t* Wp = (uint32_t*)(ws + 4096 + 602112);             // 1.18 MB packed W

    v2_kernel<<<(D + 255) / 256, 256, 0, stream>>>(W, a, v2);
    packW_kernel<<<(NTILES * KTILES * 256) / 256, 256, 0, stream>>>(W, Wp);
    attn_e_kernel<<<(N_ROWS + 7) / 8, 256, 0, stream>>>(F1, F2, F3, v2, e);

    int shmem = BM * LDS_STR * 2 + BM * 3 * 4 + 16;             // ~100 KB
    hipFuncSetAttribute(reinterpret_cast<const void*>(fused_wh_gemm_kernel),
                        hipFuncAttributeMaxDynamicSharedMemorySize, shmem);
    fused_wh_gemm_kernel<<<(N_ROWS + BM - 1) / BM, 256, shmem, stream>>>(
        F0, F1, F2, F3, e, Wp, out);
}
